// SurfaceAbstractionCD_sa1_16106127360155
// MI455X (gfx1250) — matrix operand order repaired and checked by reading, not compiled, not run on hardware
//
#include <hip/hip_runtime.h>
#include <hip/hip_bf16.h>

typedef _Float16 v16h __attribute__((ext_vector_type(16)));
typedef float    v8f  __attribute__((ext_vector_type(8)));

#define NPT   1024
#define NPTS  8192
#define NB    8
#define NS    32
#define CFEAT 6
#define MTOT  (NB*NPT*NS)   /* 262144 rows */
#define RAD2  (0.2f*0.2f)

/* ---- workspace layout (bytes) ---- */
#define OFF_X0   ((size_t)0)                       /* 262144*32*2  = 16 MB  */
#define OFF_Y0   ((size_t)16777216)                /* 262144*128*2 = 64 MB  */
#define OFF_Y1   ((size_t)83886080)                /* 262144*64*2  = 32 MB  */
#define OFF_Y2   ((size_t)0)                       /* overlays dead X0+Y0   */
#define OFF_S    ((size_t)117440512)
#define OFF_FPSI (OFF_S)                           /* 8192 int   */
#define OFF_NC   (OFF_FPSI + 32768)                /* 24576 f32  */
#define OFF_NN   (OFF_NC   + 98304)                /* 24576 f32  */
#define OFF_BIDX (OFF_NN   + 98304)                /* 262144 int */
#define OFF_W0   (OFF_BIDX + 1048576)              /* 128x32 f16 */
#define OFF_W1   (OFF_W0   + 8192)                 /* 64x64 f16  */
#define OFF_W2   (OFF_W1   + 8192)                 /* 128x64 f16 */
#define OFF_SUM0 (OFF_W2   + 16384)                /* 128 f32 */
#define OFF_SQ0  (OFF_SUM0 + 512)
#define OFF_SUM1 (OFF_SQ0  + 512)                  /* 64 */
#define OFF_SQ1  (OFF_SUM1 + 256)
#define OFF_SUM2 (OFF_SQ1  + 256)                  /* 128 */
#define OFF_SQ2  (OFF_SUM2 + 512)
#define OFF_A0   (OFF_SQ2  + 512)
#define OFF_B0   (OFF_A0   + 512)
#define OFF_A1   (OFF_B0   + 512)
#define OFF_B1   (OFF_A1   + 256)
#define OFF_A2   (OFF_B1   + 256)
#define OFF_B2   (OFF_A2   + 512)

union AFrag { v16h v; unsigned u[8]; };
union PK    { unsigned u; _Float16 h[2]; };

__device__ __forceinline__ v8f wmma16(v16h a, v16h b, v8f c) {
  return __builtin_amdgcn_wmma_f32_16x16x32_f16(false, a, false, b, (short)0, c, false, false);
}

/* ---------------- stats init ---------------- */
__global__ void init_stats_kernel(float* p) {
  for (int i = threadIdx.x; i < 640; i += 256) p[i] = 0.f;
}

/* ---------------- weight prep (f32 -> f16, [N][K] layout) ---------------- */
__global__ void prep_weights_kernel(const float* wl0, const float* wf0,
                                    const float* w1, const float* w2,
                                    _Float16* w0c, _Float16* w1h, _Float16* w2h) {
  int tid = threadIdx.x;
  for (int i = tid; i < 128 * 32; i += 256) {        /* combined loc|ftr, K padded 12->32 */
    int o = i >> 5, k = i & 31;
    float v = 0.f;
    if (o < 64) { if (k < 3) v = wl0[o * 3 + k]; }
    else        { if (k >= 3 && k < 12) v = wf0[(o - 64) * 9 + (k - 3)]; }
    w0c[i] = (_Float16)v;
  }
  for (int i = tid; i < 64 * 64;  i += 256) w1h[i] = (_Float16)w1[i];
  for (int i = tid; i < 128 * 64; i += 256) w2h[i] = (_Float16)w2[i];
}

/* ---------------- farthest point sampling: one block per batch ---------------- */
__global__ __launch_bounds__(1024) void fps_kernel(const float* center, const float* normal,
                                                   int* fidx, float* nc, float* nn, float* out) {
  int b = blockIdx.x, t = threadIdx.x;
  const float* cb = center + (size_t)b * NPTS * 3;
  const float* nb = normal + (size_t)b * NPTS * 3;
  float px[8], py[8], pz[8], tmp[8];
#pragma unroll
  for (int i = 0; i < 8; i++) {
    int p = t + i * 1024;
    px[i] = cb[p * 3]; py[i] = cb[p * 3 + 1]; pz[i] = cb[p * 3 + 2];
    tmp[i] = 1e10f;
  }
  __shared__ float s_val[32]; __shared__ int s_idx[32];
  __shared__ int s_far;       __shared__ float s_cur[3];
  int far = 0;
  float* oc = out +          (size_t)b * 3 * NPT;
  float* on = out + 24576 +  (size_t)b * 3 * NPT;
  for (int j = 0; j < NPT; j++) {
    if (t == 0) {
      fidx[b * NPT + j] = far;
      float cx = cb[far*3], cy = cb[far*3+1], cz = cb[far*3+2];
      float nx = nb[far*3], ny = nb[far*3+1], nz = nb[far*3+2];
      int o = (b * NPT + j) * 3;
      nc[o] = cx; nc[o+1] = cy; nc[o+2] = cz;
      nn[o] = nx; nn[o+1] = ny; nn[o+2] = nz;
      oc[j] = cx; oc[NPT + j] = cy; oc[2*NPT + j] = cz;
      on[j] = nx; on[NPT + j] = ny; on[2*NPT + j] = nz;
      s_cur[0] = cx; s_cur[1] = cy; s_cur[2] = cz;
    }
    __syncthreads();
    float cx = s_cur[0], cy = s_cur[1], cz = s_cur[2];
    float best = -1.f; int bi = 0;
#pragma unroll
    for (int i = 0; i < 8; i++) {
      float dx = px[i]-cx, dy = py[i]-cy, dz = pz[i]-cz;
      float d = dx*dx + dy*dy + dz*dz;
      tmp[i] = fminf(tmp[i], d);
      if (tmp[i] > best) { best = tmp[i]; bi = t + i * 1024; }
    }
#pragma unroll
    for (int off = 16; off > 0; off >>= 1) {
      float ov = __shfl_down(best, off);
      int   oi = __shfl_down(bi,   off);
      if (ov > best || (ov == best && oi < bi)) { best = ov; bi = oi; }
    }
    if ((t & 31) == 0) { s_val[t >> 5] = best; s_idx[t >> 5] = bi; }
    __syncthreads();
    if (t < 32) {
      best = s_val[t]; bi = s_idx[t];
#pragma unroll
      for (int off = 16; off > 0; off >>= 1) {
        float ov = __shfl_down(best, off);
        int   oi = __shfl_down(bi,   off);
        if (ov > best || (ov == best && oi < bi)) { best = ov; bi = oi; }
      }
      if (t == 0) s_far = bi;
    }
    __syncthreads();
    far = s_far;
  }
}

/* ---------------- ball query: one wave32 per query, ordered compaction ---------------- */
__global__ __launch_bounds__(128) void ball_query_kernel(const float* center, const float* nc, int* bidx) {
  int wv = threadIdx.x >> 5, lane = threadIdx.x & 31;
  int q = blockIdx.x * 4 + wv;           /* 0..8191 */
  int b = q >> 10;
  const float* base = center + (size_t)b * NPTS * 3;
  float qx = nc[q*3], qy = nc[q*3+1], qz = nc[q*3+2];
  int cnt = 0, slot = 0, filled = 0;
  for (int c = 0; c < NPTS / 32 && cnt < NS; ++c) {
    int p = c * 32 + lane;
    float dx = base[p*3]-qx, dy = base[p*3+1]-qy, dz = base[p*3+2]-qz;
    bool in = (dx*dx + dy*dy + dz*dz) < RAD2;
    unsigned m = (unsigned)__ballot(in);
    int nm = __popc(m);
    if (!filled && lane >= cnt && lane < cnt + nm) {
      int j = lane - cnt; unsigned mm = m;
      for (int t2 = 0; t2 < j; t2++) mm &= (mm - 1);
      slot = c * 32 + __builtin_ctz(mm);
      filled = 1;
    }
    cnt += nm;
  }
  int first = __shfl(slot, 0);
  int f0    = __shfl(filled, 0);
  int val = filled ? slot : (f0 ? first : 0);
  bidx[(size_t)q * NS + lane] = val;
}

/* ---------------- gather grouped features -> X0 (f16, K padded to 32) ---------------- */
__global__ __launch_bounds__(256) void gather_x0_kernel(const float* center, const float* normal,
                                                        const float* feature, const float* nc,
                                                        const int* bidx, _Float16* X0) {
  int r = blockIdx.x * 256 + threadIdx.x;          /* row 0..262143 */
  int b = r >> 15; int rem = r & 32767; int m = rem >> 5;
  int pt = bidx[r];
  const float* c3 = center + ((size_t)b * NPTS + pt) * 3;
  const float* n3 = normal + ((size_t)b * NPTS + pt) * 3;
  const float* q3 = nc + (size_t)(b * NPT + m) * 3;
  const float* fb = feature + (size_t)b * CFEAT * NPTS + pt;
  float v[12];
  v[0] = c3[0]-q3[0]; v[1] = c3[1]-q3[1]; v[2] = c3[2]-q3[2];
  v[3] = n3[0]; v[4] = n3[1]; v[5] = n3[2];
#pragma unroll
  for (int k = 0; k < 6; k++) v[6 + k] = fb[(size_t)k * NPTS];
  unsigned o16[16];
#pragma unroll
  for (int i = 0; i < 16; i++) {
    PK pk;
    pk.h[0] = (_Float16)((2*i     < 12) ? v[2*i]     : 0.f);
    pk.h[1] = (_Float16)((2*i + 1 < 12) ? v[2*i + 1] : 0.f);
    o16[i] = pk.u;
  }
  uint4* dst = (uint4*)(X0 + (size_t)r * 32);
#pragma unroll
  for (int k = 0; k < 4; k++) {
    uint4 t; t.x = o16[4*k]; t.y = o16[4*k+1]; t.z = o16[4*k+2]; t.w = o16[4*k+3];
    dst[k] = t;
  }
}

/* ---------------- GEMM0: X0(262144x32) x W0^T(32x128) -> Y0 + stats ---------------- */
__global__ __launch_bounds__(256) void gemm0_kernel(const _Float16* X0, const _Float16* w,
                                                    _Float16* y0, float* gsum, float* gsq) {
  __shared__ _Float16 sw[128 * 32];
  __shared__ float sacc[128], sacc2[128];
  int tid = threadIdx.x;
  { const unsigned* s = (const unsigned*)w; unsigned* d = (unsigned*)sw;
    for (int i = tid; i < 2048; i += 256) d[i] = s[i]; }
  if (tid < 128) { sacc[tid] = 0.f; sacc2[tid] = 0.f; }
  __syncthreads();
  int wv = tid >> 5, lane = tid & 31;
  int m0 = blockIdx.x * 128 + wv * 16;
  int rowA = m0 + (lane & 15);
  AFrag a;
  const unsigned* xp = (const unsigned*)(X0 + (size_t)rowA * 32);
  int kd = (lane < 16) ? 0 : 4;
#pragma unroll
  for (int j = 0; j < 8; j++) a.u[j] = xp[j + kd + ((j < 4) ? 0 : 4)];
  v8f z8 = {0.f,0.f,0.f,0.f,0.f,0.f,0.f,0.f};
  v8f cac[8];
#pragma unroll
  for (int n = 0; n < 8; n++) cac[n] = z8;
  int colb = lane & 15, kb2 = (lane < 16) ? 0 : 16;
#pragma unroll
  for (int n = 0; n < 8; n++) {
    AFrag bf;
    const unsigned* wp = (const unsigned*)(sw + (n * 16 + colb) * 32 + kb2);
#pragma unroll
    for (int j = 0; j < 8; j++) bf.u[j] = wp[j + ((j < 4) ? 0 : 4) - (kb2 >> 2)];
    cac[n] = wmma16(a.v, bf.v, cac[n]);
  }
  int rb = (lane < 16) ? 0 : 8;
#pragma unroll
  for (int n = 0; n < 8; n++) {
    int col = n * 16 + colb;
    float s = 0.f, s2 = 0.f;
#pragma unroll
    for (int r = 0; r < 8; r++) {
      float vv = cac[n][r];
      y0[(size_t)(m0 + rb + r) * 128 + col] = (_Float16)vv;
      s += vv; s2 += vv * vv;
    }
    atomicAdd(&sacc[col], s); atomicAdd(&sacc2[col], s2);
  }
  __syncthreads();
  if (tid < 128) { atomicAdd(&gsum[tid], sacc[tid]); atomicAdd(&gsq[tid], sacc2[tid]); }
}

/* ---------------- fold BN stats into affine a,b ---------------- */
__global__ void finalize_kernel(const float* gsum, const float* gsq,
                                const float* g, const float* be,
                                float* A, float* Bc, int nch) {
  int c = threadIdx.x;
  if (c < nch) {
    const float invN = 1.0f / (float)MTOT;
    float mean = gsum[c] * invN;
    float var = gsq[c] * invN - mean * mean;
    float inv = rsqrtf(var + 1e-5f);
    float a = g[c] * inv;
    A[c] = a; Bc[c] = be[c] - mean * a;
  }
}

/* ---------------- GEMM1: relu(bn(loc)+bn(ftr)) fused into A; 64->64 ---------------- */
__global__ __launch_bounds__(256) void gemm1_kernel(const _Float16* y0, const _Float16* w,
                                                    const float* a0, const float* b0,
                                                    _Float16* y1, float* gsum, float* gsq) {
  __shared__ _Float16 sw[64 * 64];
  __shared__ float sA[128], sB[128];
  __shared__ float sacc[64], sacc2[64];
  int tid = threadIdx.x;
  { const unsigned* s = (const unsigned*)w; unsigned* d = (unsigned*)sw;
    for (int i = tid; i < 2048; i += 256) d[i] = s[i]; }
  if (tid < 128) { sA[tid] = a0[tid]; sB[tid] = b0[tid]; }
  if (tid < 64)  { sacc[tid] = 0.f; sacc2[tid] = 0.f; }
  __syncthreads();
  int wv = tid >> 5, lane = tid & 31;
  int m0 = blockIdx.x * 128 + wv * 16;
  int rowA = m0 + (lane & 15);
  const _Float16* yr = y0 + (size_t)rowA * 128;
  int kd = (lane < 16) ? 0 : 4;
  AFrag a[2];
#pragma unroll
  for (int kc = 0; kc < 2; kc++)
#pragma unroll
    for (int j = 0; j < 8; j++) {
      int c0 = kc * 32 + 2 * (j + kd + ((j < 4) ? 0 : 4));
      PK lp, fp, o;
      lp.u = *(const unsigned*)(yr + c0);
      fp.u = *(const unsigned*)(yr + 64 + c0);
      float z0 = (float)lp.h[0]*sA[c0]   + sB[c0]   + (float)fp.h[0]*sA[64+c0]   + sB[64+c0];
      float z1 = (float)lp.h[1]*sA[c0+1] + sB[c0+1] + (float)fp.h[1]*sA[64+c0+1] + sB[64+c0+1];
      o.h[0] = (_Float16)fmaxf(z0, 0.f); o.h[1] = (_Float16)fmaxf(z1, 0.f);
      a[kc].u[j] = o.u;
    }
  v8f z8 = {0.f,0.f,0.f,0.f,0.f,0.f,0.f,0.f};
  v8f cac[4];
#pragma unroll
  for (int n = 0; n < 4; n++) cac[n] = z8;
  int colb = lane & 15, kb2 = (lane < 16) ? 0 : 16;
#pragma unroll
  for (int n = 0; n < 4; n++)
#pragma unroll
    for (int kc = 0; kc < 2; kc++) {
      AFrag bf;
      const unsigned* wp = (const unsigned*)(sw + (n * 16 + colb) * 64 + kc * 32 + kb2);
#pragma unroll
      for (int j = 0; j < 8; j++) bf.u[j] = wp[j + ((j < 4) ? 0 : 4) - (kb2 >> 2)];
      cac[n] = wmma16(a[kc].v, bf.v, cac[n]);
    }
  int rb = (lane < 16) ? 0 : 8;
#pragma unroll
  for (int n = 0; n < 4; n++) {
    int col = n * 16 + colb;
    float s = 0.f, s2 = 0.f;
#pragma unroll
    for (int r = 0; r < 8; r++) {
      float vv = cac[n][r];
      y1[(size_t)(m0 + rb + r) * 64 + col] = (_Float16)vv;
      s += vv; s2 += vv * vv;
    }
    atomicAdd(&sacc[col], s); atomicAdd(&sacc2[col], s2);
  }
  __syncthreads();
  if (tid < 64) { atomicAdd(&gsum[tid], sacc[tid]); atomicAdd(&gsq[tid], sacc2[tid]); }
}

/* ---------------- GEMM2: relu(bn(y1)) fused into A; 64->128 ---------------- */
__global__ __launch_bounds__(256) void gemm2_kernel(const _Float16* y1, const _Float16* w,
                                                    const float* a1, const float* b1,
                                                    _Float16* y2, float* gsum, float* gsq) {
  __shared__ _Float16 sw[128 * 64];
  __shared__ float sA[64], sB[64];
  __shared__ float sacc[128], sacc2[128];
  int tid = threadIdx.x;
  { const unsigned* s = (const unsigned*)w; unsigned* d = (unsigned*)sw;
    for (int i = tid; i < 4096; i += 256) d[i] = s[i]; }
  if (tid < 64)  { sA[tid] = a1[tid]; sB[tid] = b1[tid]; }
  if (tid < 128) { sacc[tid] = 0.f; sacc2[tid] = 0.f; }
  __syncthreads();
  int wv = tid >> 5, lane = tid & 31;
  int m0 = blockIdx.x * 128 + wv * 16;
  int rowA = m0 + (lane & 15);
  const _Float16* yr = y1 + (size_t)rowA * 64;
  int kd = (lane < 16) ? 0 : 4;
  AFrag a[2];
#pragma unroll
  for (int kc = 0; kc < 2; kc++)
#pragma unroll
    for (int j = 0; j < 8; j++) {
      int c0 = kc * 32 + 2 * (j + kd + ((j < 4) ? 0 : 4));
      PK p, o;
      p.u = *(const unsigned*)(yr + c0);
      float z0 = (float)p.h[0]*sA[c0]   + sB[c0];
      float z1 = (float)p.h[1]*sA[c0+1] + sB[c0+1];
      o.h[0] = (_Float16)fmaxf(z0, 0.f); o.h[1] = (_Float16)fmaxf(z1, 0.f);
      a[kc].u[j] = o.u;
    }
  v8f z8 = {0.f,0.f,0.f,0.f,0.f,0.f,0.f,0.f};
  v8f cac[8];
#pragma unroll
  for (int n = 0; n < 8; n++) cac[n] = z8;
  int colb = lane & 15, kb2 = (lane < 16) ? 0 : 16;
#pragma unroll
  for (int n = 0; n < 8; n++)
#pragma unroll
    for (int kc = 0; kc < 2; kc++) {
      AFrag bf;
      const unsigned* wp = (const unsigned*)(sw + (n * 16 + colb) * 64 + kc * 32 + kb2);
#pragma unroll
      for (int j = 0; j < 8; j++) bf.u[j] = wp[j + ((j < 4) ? 0 : 4) - (kb2 >> 2)];
      cac[n] = wmma16(a[kc].v, bf.v, cac[n]);
    }
  int rb = (lane < 16) ? 0 : 8;
#pragma unroll
  for (int n = 0; n < 8; n++) {
    int col = n * 16 + colb;
    float s = 0.f, s2 = 0.f;
#pragma unroll
    for (int r = 0; r < 8; r++) {
      float vv = cac[n][r];
      y2[(size_t)(m0 + rb + r) * 128 + col] = (_Float16)vv;
      s += vv; s2 += vv * vv;
    }
    atomicAdd(&sacc[col], s); atomicAdd(&sacc2[col], s2);
  }
  __syncthreads();
  if (tid < 128) { atomicAdd(&gsum[tid], sacc[tid]); atomicAdd(&gsq[tid], sacc2[tid]); }
}

/* ---------------- final: relu(bn2(y2)), max over 32 samples, transpose out ---------------- */
__global__ __launch_bounds__(256) void final_max_kernel(const _Float16* y2, const float* a2,
                                                        const float* b2, float* out) {
  int wv = threadIdx.x >> 5, lane = threadIdx.x & 31;
  int gidx = blockIdx.x * 8 + wv;    /* 0..8191 = (b,m) */
  int b = gidx >> 10, m = gidx & 1023;
  int c0 = lane * 4;
  float A[4], Bc[4], mx[4];
#pragma unroll
  for (int i = 0; i < 4; i++) { A[i] = a2[c0+i]; Bc[i] = b2[c0+i]; mx[i] = 0.f; }
  const _Float16* base = y2 + (size_t)gidx * 32 * 128 + c0;
  for (int s = 0; s < 32; s++) {
    union { uint2 u; _Float16 h[4]; } v;
    v.u = *(const uint2*)(base + (size_t)s * 128);
#pragma unroll
    for (int i = 0; i < 4; i++) {
      float z = (float)v.h[i] * A[i] + Bc[i];
      mx[i] = fmaxf(mx[i], fmaxf(z, 0.f));
    }
  }
  float* o = out + 49152 + (size_t)b * 131072 + (size_t)c0 * 1024 + m;
#pragma unroll
  for (int i = 0; i < 4; i++) o[(size_t)i * 1024] = mx[i];
}

extern "C" void kernel_launch(void* const* d_in, const int* in_sizes, int n_in,
                              void* d_out, int out_size, void* d_ws, size_t ws_size,
                              hipStream_t stream) {
  (void)in_sizes; (void)n_in; (void)out_size; (void)ws_size;
  const float* center  = (const float*)d_in[0];
  const float* normal  = (const float*)d_in[1];
  const float* feature = (const float*)d_in[2];
  const float* w_l0 = (const float*)d_in[3];
  const float* g_l0 = (const float*)d_in[5];
  const float* be_l0= (const float*)d_in[6];
  const float* w_f0 = (const float*)d_in[7];
  const float* g_f0 = (const float*)d_in[9];
  const float* be_f0= (const float*)d_in[10];
  const float* w1   = (const float*)d_in[11];
  const float* g1   = (const float*)d_in[13];
  const float* be1  = (const float*)d_in[14];
  const float* w2   = (const float*)d_in[15];
  const float* g2   = (const float*)d_in[17];
  const float* be2  = (const float*)d_in[18];

  char* ws = (char*)d_ws;
  _Float16* X0 = (_Float16*)(ws + OFF_X0);
  _Float16* Y0 = (_Float16*)(ws + OFF_Y0);
  _Float16* Y1 = (_Float16*)(ws + OFF_Y1);
  _Float16* Y2 = (_Float16*)(ws + OFF_Y2);
  int*   fidx = (int*)(ws + OFF_FPSI);
  float* nc   = (float*)(ws + OFF_NC);
  float* nn   = (float*)(ws + OFF_NN);
  int*   bidx = (int*)(ws + OFF_BIDX);
  _Float16* W0 = (_Float16*)(ws + OFF_W0);
  _Float16* W1 = (_Float16*)(ws + OFF_W1);
  _Float16* W2 = (_Float16*)(ws + OFF_W2);
  float* SUM0 = (float*)(ws + OFF_SUM0); float* SQ0 = (float*)(ws + OFF_SQ0);
  float* SUM1 = (float*)(ws + OFF_SUM1); float* SQ1 = (float*)(ws + OFF_SQ1);
  float* SUM2 = (float*)(ws + OFF_SUM2); float* SQ2 = (float*)(ws + OFF_SQ2);
  float* A0 = (float*)(ws + OFF_A0); float* B0 = (float*)(ws + OFF_B0);
  float* A1 = (float*)(ws + OFF_A1); float* B1 = (float*)(ws + OFF_B1);
  float* A2 = (float*)(ws + OFF_A2); float* B2 = (float*)(ws + OFF_B2);
  float* out = (float*)d_out;

  init_stats_kernel<<<1, 256, 0, stream>>>(SUM0);
  prep_weights_kernel<<<1, 256, 0, stream>>>(w_l0, w_f0, w1, w2, W0, W1, W2);
  fps_kernel<<<NB, 1024, 0, stream>>>(center, normal, fidx, nc, nn, out);
  ball_query_kernel<<<(NB * NPT) / 4, 128, 0, stream>>>(center, nc, bidx);
  gather_x0_kernel<<<MTOT / 256, 256, 0, stream>>>(center, normal, feature, nc, bidx, X0);
  gemm0_kernel<<<MTOT / 128, 256, 0, stream>>>(X0, W0, Y0, SUM0, SQ0);
  finalize_kernel<<<1, 64, 0, stream>>>(SUM0,      SQ0,      g_l0, be_l0, A0,      B0,      64);
  finalize_kernel<<<1, 64, 0, stream>>>(SUM0 + 64, SQ0 + 64, g_f0, be_f0, A0 + 64, B0 + 64, 64);
  gemm1_kernel<<<MTOT / 128, 256, 0, stream>>>(Y0, W1, A0, B0, Y1, SUM1, SQ1);
  finalize_kernel<<<1, 64, 0, stream>>>(SUM1, SQ1, g1, be1, A1, B1, 64);
  gemm2_kernel<<<MTOT / 128, 256, 0, stream>>>(Y1, W2, A1, B1, Y2, SUM2, SQ2);
  finalize_kernel<<<1, 128, 0, stream>>>(SUM2, SQ2, g2, be2, A2, B2, 128);
  final_max_kernel<<<(NB * NPT) / 8, 256, 0, stream>>>(Y2, A2, B2, out);
}
